// SwinTransformerBlock_30863634989661
// MI455X (gfx1250) — compile-verified, matrix operand order checked by reading, not run on hardware
//
#include <hip/hip_runtime.h>
#include <math.h>

// ---------------------------------------------------------------------------
// Swin-3D block for MI455X (gfx1250): all GEMMs via v_wmma_f32_16x16x32_bf16,
// depthwise-conv weights staged via async global->LDS copies (overlapped with
// the WMMA GEMM), V kept channel-major so every WMMA fragment load is a pair
// of contiguous b128 loads.
// ---------------------------------------------------------------------------

#define DEV static __device__ __forceinline__

typedef __attribute__((ext_vector_type(16))) __bf16 v16bf;
typedef __attribute__((ext_vector_type(8)))  float  v8f;
typedef __attribute__((ext_vector_type(4)))  int    v4i;
typedef __attribute__((address_space(1))) v4i* gv4i_p;   // global int4*
typedef __attribute__((address_space(3))) v4i* lv4i_p;   // LDS int4*

union Frag16 { unsigned short s[16]; v16bf v; };

constexpr int   C_   = 96;
constexpr int   RESX = 56;
constexpr int   VOL  = RESX * RESX * RESX;   // 175616 tokens
constexpr int   NT   = 343;                  // tokens per 7x7x7 window
constexpr int   NTP  = 352;                  // padded to 22 * 16
constexpr int   NWIN = 512;                  // 8^3 windows
constexpr int   WROW = NTP * C_;             // 33792 elems per window plane
constexpr int   HID  = 384;
constexpr float EPSV = 1e-5f;
constexpr float QKSC = 0.1020620726159658f;  // 1/sqrt(96)

#if defined(__has_builtin)
#if __has_builtin(__builtin_amdgcn_global_load_async_to_lds_b128) && \
    __has_builtin(__builtin_amdgcn_s_wait_asynccnt)
#define USE_ASYNC_LDS 1
#endif
#endif
#ifndef USE_ASYNC_LDS
#define USE_ASYNC_LDS 0
#endif

DEV unsigned short f2bf(float f) {             // f32 -> bf16 RNE
  union { float f; unsigned u; } v; v.f = f;
  return (unsigned short)((v.u + 0x7FFFu + ((v.u >> 16) & 1u)) >> 16);
}

DEV float wsum(float v) {                      // wave32 butterfly sum
#pragma unroll
  for (int m = 16; m >= 1; m >>= 1) v += __shfl_xor(v, m, 32);
  return v;
}

// A/B fragment where element(r,k) = base[r*ld + k]
// (A row-major over rows r; B stored [n][k] with r = column). Per lane this is
// two contiguous 16-byte runs -> compiler emits 2x global_load_b128 / ds reads.
DEV v16bf fragRK(const unsigned short* base, int ld) {
  int lane = threadIdx.x & 31;
  int half = lane >> 4, r = lane & 15;
  Frag16 f;
#pragma unroll
  for (int j = 0; j < 16; ++j) {
    int k = ((j >> 3) << 4) + (half << 3) + (j & 7);
    f.s[j] = base[r * ld + k];
  }
  return f.v;
}

DEV v8f wmma_bf16(v16bf a, v16bf b, v8f c) {
  return __builtin_amdgcn_wmma_f32_16x16x32_bf16(false, a, false, b,
                                                 (short)0, c, false, false);
}

DEV int region1(int g) { return g < 49 ? 0 : (g < 53 ? 1 : 2); }

// ---------------------------------------------------------------------------
// Kernel A: LN1 + roll(-3,-3,-3) + window partition -> bf16 [win][352][96]
// one wave per token, 3 channels per lane
// ---------------------------------------------------------------------------
__global__ void k_ln1_window(const float* __restrict__ x,
                             const float* __restrict__ g,
                             const float* __restrict__ b,
                             unsigned short* __restrict__ xw) {
  int wave = threadIdx.x >> 5, lane = threadIdx.x & 31;
  int tp  = blockIdx.x * 8 + wave;             // [0, NWIN*NTP)
  int win = tp / NTP, n = tp % NTP;
  size_t ob = (size_t)win * WROW + (size_t)n * C_;
  if (n >= NT) {                               // zero the pad rows
#pragma unroll
    for (int j = 0; j < 3; ++j) xw[ob + lane + 32 * j] = 0;
    return;
  }
  int ls = n / 49, lh = (n % 49) / 7, lw = n % 7;
  int wd = win >> 6, wh = (win >> 3) & 7, ww = win & 7;
  int s = (wd * 7 + ls + 3) % RESX;
  int h = (wh * 7 + lh + 3) % RESX;
  int w = (ww * 7 + lw + 3) % RESX;
  int sp = (s * RESX + h) * RESX + w;
  float vals[3], acc = 0.f;
#pragma unroll
  for (int j = 0; j < 3; ++j) {
    vals[j] = x[(size_t)(lane + 32 * j) * VOL + sp];
    acc += vals[j];
  }
  float mean = wsum(acc) * (1.f / 96.f);
  float var = 0.f;
#pragma unroll
  for (int j = 0; j < 3; ++j) { float d = vals[j] - mean; var += d * d; }
  var = wsum(var) * (1.f / 96.f);
  float inv = rsqrtf(var + EPSV);
#pragma unroll
  for (int j = 0; j < 3; ++j) {
    int c = lane + 32 * j;
    xw[ob + c] = f2bf((vals[j] - mean) * inv * g[c] + b[c]);
  }
}

// ---------------------------------------------------------------------------
// fp32 -> bf16 converter for weight matrices
// ---------------------------------------------------------------------------
__global__ void k_cvt_bf16(const float* __restrict__ src,
                           unsigned short* __restrict__ dst, int n) {
  int i = blockIdx.x * 256 + threadIdx.x;
  if (i < n) dst[i] = f2bf(src[i]);
}

// ---------------------------------------------------------------------------
// Kernel B: per-window QKV = dwconv3x3x3( W(96x96) @ X(96x343) + bias ) + dwb
// one workgroup (8 waves) per window; 32-channel chunks staged in LDS.
// Depthwise weights are copied global->LDS with async-LDS loads overlapped
// with the WMMA GEMM.  Q,K stored token-major [n][96]; V channel-major
// [96][352] so the attention P@V B-fragments are contiguous.
// ---------------------------------------------------------------------------
__global__ void k_qkv(const unsigned short* __restrict__ xw,
                      const unsigned short* __restrict__ wqb,
                      const unsigned short* __restrict__ wkb,
                      const unsigned short* __restrict__ wvb,
                      const float* __restrict__ bq, const float* __restrict__ bk,
                      const float* __restrict__ bv,
                      const float* __restrict__ dqw, const float* __restrict__ dkw,
                      const float* __restrict__ dvw,
                      const float* __restrict__ dqb, const float* __restrict__ dkb,
                      const float* __restrict__ dvb,
                      unsigned short* __restrict__ qb,
                      unsigned short* __restrict__ kb,
                      unsigned short* __restrict__ vb) {
  __shared__ float proj[32 * NTP];             // 45 KB
  __shared__ float dwW[32 * 27];               // 3.4 KB (one channel chunk)
  int win = blockIdx.x;
  int tid = threadIdx.x, wv = tid >> 5, lane = tid & 31;
  int half = lane >> 4, r = lane & 15;
  size_t wbase = (size_t)win * WROW;
  __builtin_prefetch(wqb, 0, 1);
  __builtin_prefetch(wkb, 0, 1);
  __builtin_prefetch(wvb, 0, 1);
  // zero the pad regions of q/k (rows) and v (columns) once
  for (int e = tid; e < 9 * C_; e += 256) {
    int j = e / C_, c = e % C_;
    size_t o = wbase + (size_t)(NT + j) * C_ + c;
    qb[o] = 0; kb[o] = 0;
    vb[wbase + (size_t)c * NTP + (NT + j)] = 0;
  }
  for (int p = 0; p < 3; ++p) {
    const unsigned short* W = p == 0 ? wqb : (p == 1 ? wkb : wvb);
    const float* cb = p == 0 ? bq : (p == 1 ? bk : bv);
    const float* dw = p == 0 ? dqw : (p == 1 ? dkw : dvw);
    const float* db = p == 0 ? dqb : (p == 1 ? dkb : dvb);
    unsigned short* out = p == 0 ? qb : (p == 1 ? kb : vb);
    for (int ch = 0; ch < 3; ++ch) {
      int c0 = ch * 32;
      __syncthreads();                         // proj/dwW reuse fence
      // kick off async copy of this chunk's depthwise weights into LDS;
      // it overlaps with the WMMA GEMM below.
      {
        const float* src = dw + c0 * 27;       // 864 floats = 216 x b128
#if USE_ASYNC_LDS
        for (int e = tid; e < (32 * 27) / 4; e += 256)
          __builtin_amdgcn_global_load_async_to_lds_b128(
              (gv4i_p)(src + e * 4), (lv4i_p)(dwW + e * 4), 0, 0);
#else
        for (int e = tid; e < 32 * 27; e += 256) dwW[e] = src[e];
#endif
      }
      // hoist the 6 weight A-fragments (2 m-tiles x 3 k-steps) for this chunk
      v16bf aW[2][3];
#pragma unroll
      for (int mt = 0; mt < 2; ++mt)
#pragma unroll
        for (int ks = 0; ks < 3; ++ks)
          aW[mt][ks] = fragRK(W + (c0 + mt * 16) * C_ + ks * 32, C_);
      for (int t = wv; t < 44; t += 8) {       // 2 m-tiles x 22 n-tiles
        int mt = t / 22, nt = t % 22;
        int cm = c0 + mt * 16;
        v8f acc;
#pragma unroll
        for (int i = 0; i < 8; ++i) acc[i] = cb[cm + i + 8 * half];  // 1x1 bias
#pragma unroll
        for (int ks = 0; ks < 3; ++ks) {
          v16bf bf = fragRK(xw + wbase + (size_t)nt * 16 * C_ + ks * 32, C_);
          acc = wmma_bf16(aW[mt][ks], bf, acc);
        }
        int col = nt * 16 + r;
#pragma unroll
        for (int i = 0; i < 8; ++i)
          proj[(mt * 16 + i + 8 * half) * NTP + col] = acc[i];
      }
#if USE_ASYNC_LDS
      __builtin_amdgcn_s_wait_asynccnt(0);
#endif
      __syncthreads();
      // depthwise 3x3x3 (zero pad) within the 7^3 window, weights from LDS
      for (int e = tid; e < 32 * NT; e += 256) {
        int cl = e / NT, n = e % NT;
        int c = c0 + cl;
        int z = n / 49, y = (n % 49) / 7, xv = n % 7;
        float a = db[c];
        const float* wgt = dwW + cl * 27;
#pragma unroll
        for (int dz = 0; dz < 3; ++dz) {
          int zz = z + dz - 1; if ((unsigned)zz >= 7u) continue;
#pragma unroll
          for (int dy = 0; dy < 3; ++dy) {
            int yy = y + dy - 1; if ((unsigned)yy >= 7u) continue;
#pragma unroll
            for (int dx = 0; dx < 3; ++dx) {
              int xx = xv + dx - 1; if ((unsigned)xx >= 7u) continue;
              a += wgt[dz * 9 + dy * 3 + dx] * proj[cl * NTP + (zz * 49 + yy * 7 + xx)];
            }
          }
        }
        unsigned short bfv = f2bf(a);
        if (p == 2) out[wbase + (size_t)c * NTP + n] = bfv;   // V: [c][352]
        else        out[wbase + (size_t)n * C_ + c] = bfv;    // Q,K: [n][96]
      }
    }
  }
}

// ---------------------------------------------------------------------------
// Kernel C: per (window, 16-row tile) attention.
// scores = QK^T/sqrt(C) + analytic shift-mask; softmax; out = P @ V
// 4 waves per workgroup; mask derived from region IDs (static-shape function)
// ---------------------------------------------------------------------------
__global__ void k_attn(const unsigned short* __restrict__ qb,
                       const unsigned short* __restrict__ kb,
                       const unsigned short* __restrict__ vb,
                       float* __restrict__ att) {
  __shared__ float sc[16 * NTP];               // 22.5 KB
  __shared__ unsigned short pb[16 * NTP];      // 11.3 KB
  int win = blockIdx.x / 22, nt0 = blockIdx.x % 22;
  int tid = threadIdx.x, wv = tid >> 5, lane = tid & 31;
  int half = lane >> 4, r = lane & 15;
  size_t wbase = (size_t)win * WROW;
  int nBase = nt0 * 16;
  int wd = win >> 6, wh = (win >> 3) & 7, ww = win & 7;

  // hoist Q fragments (shared across all m-tiles)
  v16bf aF[3];
#pragma unroll
  for (int ks = 0; ks < 3; ++ks)
    aF[ks] = fragRK(qb + wbase + (size_t)nBase * C_ + ks * 32, C_);

  for (int mt = wv; mt < 22; mt += 4) {
    v8f acc;
#pragma unroll
    for (int i = 0; i < 8; ++i) acc[i] = 0.f;
#pragma unroll
    for (int ks = 0; ks < 3; ++ks) {
      v16bf bf = fragRK(kb + wbase + (size_t)mt * 16 * C_ + ks * 32, C_);
      acc = wmma_bf16(aF[ks], bf, acc);
    }
    int mTok = mt * 16 + r;
    int mls = mTok / 49, mlh = (mTok % 49) / 7, mlw = mTok % 7;
    int mrs = region1(wd * 7 + mls), mrh = region1(wh * 7 + mlh),
        mrw = region1(ww * 7 + mlw);
    bool mOk = (mTok < NT);
#pragma unroll
    for (int i = 0; i < 8; ++i) {
      int row = i + 8 * half;
      float v;
      if (!mOk) v = -1e30f;                    // padded columns -> exp() = 0
      else {
        int nTok = nBase + row;
        int ls2 = nTok / 49, lh2 = (nTok % 49) / 7, lw2 = nTok % 7;
        int rs2 = region1(wd * 7 + ls2), rh2 = region1(wh * 7 + lh2),
            rw2 = region1(ww * 7 + lw2);
        v = acc[i] * QKSC;
        if (rs2 != mrs || rh2 != mrh || rw2 != mrw) v -= 100.f;
      }
      sc[row * NTP + mTok] = v;
    }
  }
  __syncthreads();
  {                                            // softmax: 8 threads per row
    int row = tid >> 3, sub = tid & 7;
    float mx = -1e30f;
    for (int j = sub; j < NTP; j += 8) mx = fmaxf(mx, sc[row * NTP + j]);
#pragma unroll
    for (int d = 1; d < 8; d <<= 1) mx = fmaxf(mx, __shfl_xor(mx, d, 32));
    float sm = 0.f;
    for (int j = sub; j < NTP; j += 8) {
      float e = __expf(sc[row * NTP + j] - mx);
      sc[row * NTP + j] = e; sm += e;
    }
#pragma unroll
    for (int d = 1; d < 8; d <<= 1) sm += __shfl_xor(sm, d, 32);
    float inv = 1.f / sm;
    for (int j = sub; j < NTP; j += 8)
      pb[row * NTP + j] = f2bf(sc[row * NTP + j] * inv);
  }
  __syncthreads();
  // out = P(16x352) @ V(352x96): 6 column tiles, 11 k-steps.
  // hoist all 11 P fragments; V is channel-major so B-fragments are contiguous
  v16bf aP[11];
#pragma unroll
  for (int ks = 0; ks < 11; ++ks) aP[ks] = fragRK(pb + ks * 32, NTP);
  for (int ct = wv; ct < 6; ct += 4) {
    v8f acc;
#pragma unroll
    for (int i = 0; i < 8; ++i) acc[i] = 0.f;
#pragma unroll
    for (int ks = 0; ks < 11; ++ks) {
      v16bf bf = fragRK(vb + wbase + (size_t)(ct * 16) * NTP + ks * 32, NTP);
      acc = wmma_bf16(aP[ks], bf, acc);
    }
#pragma unroll
    for (int i = 0; i < 8; ++i) {
      int row = i + 8 * half;
      att[wbase + (size_t)(nBase + row) * C_ + ct * 16 + r] = acc[i];
    }
  }
}

// ---------------------------------------------------------------------------
// Kernel D: torch-faithful (C,N) axis mix + window reverse + roll(+3) + shortcut
// ---------------------------------------------------------------------------
__global__ void k_resid(const float* __restrict__ x,
                        const float* __restrict__ att,
                        float* __restrict__ xres) {
  int gid = blockIdx.x * 256 + threadIdx.x;
  const int quarter = (VOL * C_) / 4;          // exact
#pragma unroll
  for (int j = 0; j < 4; ++j) {
    int e = gid + j * quarter;
    int t = e / C_, c = e % C_;
    int s = t / 3136, rem = t % 3136, h = rem / RESX, w = rem % RESX;
    int s2 = (s + RESX - 3) % RESX;
    int h2 = (h + RESX - 3) % RESX;
    int w2 = (w + RESX - 3) % RESX;
    int win = (s2 / 7) * 64 + (h2 / 7) * 8 + (w2 / 7);
    int n   = (s2 % 7) * 49 + (h2 % 7) * 7 + (w2 % 7);
    int idx = n * C_ + c;                      // out.transpose.reshape remap
    int m = idx % NT, cc = idx / NT;
    xres[(size_t)t * C_ + c] =
        x[(size_t)c * VOL + t] + att[(size_t)win * WROW + (size_t)m * C_ + cc];
  }
}

// ---------------------------------------------------------------------------
// Kernel E: LN2 + MLP (96 -> 384 -> 96, exact GELU) + residual, NCDHW store
// 64 tokens per workgroup (8 waves)
// ---------------------------------------------------------------------------
__global__ void k_mlp(const float* __restrict__ xres,
                      const float* __restrict__ g, const float* __restrict__ b,
                      const unsigned short* __restrict__ f1wb,
                      const float* __restrict__ f1b,
                      const unsigned short* __restrict__ f2wb,
                      const float* __restrict__ f2b,
                      float* __restrict__ y) {
  __shared__ unsigned short xln[64 * C_];      // 12 KB
  __shared__ unsigned short hid[64 * HID];     // 48 KB
  int tid = threadIdx.x, wv = tid >> 5, lane = tid & 31;
  int half = lane >> 4, r = lane & 15;
  int tBase = blockIdx.x * 64;
  __builtin_prefetch(f1wb, 0, 1);
  __builtin_prefetch(f2wb, 0, 1);
  // LN2: one wave per token, 8 tokens per wave
  for (int it = 0; it < 8; ++it) {
    int tl = wv * 8 + it;
    const float* xr = xres + (size_t)(tBase + tl) * C_;
    float vals[3], s = 0.f;
#pragma unroll
    for (int j = 0; j < 3; ++j) { vals[j] = xr[lane + 32 * j]; s += vals[j]; }
    float mean = wsum(s) * (1.f / 96.f);
    float var = 0.f;
#pragma unroll
    for (int j = 0; j < 3; ++j) { float d = vals[j] - mean; var += d * d; }
    var = wsum(var) * (1.f / 96.f);
    float inv = rsqrtf(var + EPSV);
#pragma unroll
    for (int j = 0; j < 3; ++j) {
      int c = lane + 32 * j;
      xln[tl * C_ + c] = f2bf((vals[j] - mean) * inv * g[c] + b[c]);
    }
  }
  __syncthreads();
  // GEMM1 (64x96 @ 96x384) + exact GELU
  for (int t = wv; t < 96; t += 8) {
    int mt = t / 24, nt = t % 24;
    v8f acc;
#pragma unroll
    for (int i = 0; i < 8; ++i) acc[i] = f1b[nt * 16 + r];
#pragma unroll
    for (int ks = 0; ks < 3; ++ks) {
      v16bf a  = fragRK(xln + mt * 16 * C_ + ks * 32, C_);
      v16bf bf = fragRK(f1wb + (size_t)nt * 16 * C_ + ks * 32, C_);
      acc = wmma_bf16(a, bf, acc);
    }
    int col = nt * 16 + r;
#pragma unroll
    for (int i = 0; i < 8; ++i) {
      float v = acc[i];
      v = 0.5f * v * (1.f + erff(v * 0.7071067811865475f));
      hid[(mt * 16 + i + 8 * half) * HID + col] = f2bf(v);
    }
  }
  __syncthreads();
  // GEMM2 (64x384 @ 384x96) + residual + NCDHW transpose store
  for (int t = wv; t < 24; t += 8) {
    int mt = t / 6, ct = t % 6;
    v8f acc;
#pragma unroll
    for (int i = 0; i < 8; ++i) acc[i] = f2b[ct * 16 + r];
#pragma unroll
    for (int ks = 0; ks < 12; ++ks) {
      v16bf a  = fragRK(hid + mt * 16 * HID + ks * 32, HID);
      v16bf bf = fragRK(f2wb + (size_t)ct * 16 * HID + ks * 32, HID);
      acc = wmma_bf16(a, bf, acc);
    }
    int cout = ct * 16 + r;
#pragma unroll
    for (int i = 0; i < 8; ++i) {
      int tok = tBase + mt * 16 + i + 8 * half;
      y[(size_t)cout * VOL + tok] = acc[i] + xres[(size_t)tok * C_ + cout];
    }
  }
}

// ---------------------------------------------------------------------------
extern "C" void kernel_launch(void* const* d_in, const int* in_sizes, int n_in,
                              void* d_out, int out_size, void* d_ws, size_t ws_size,
                              hipStream_t stream) {
  (void)in_sizes; (void)n_in; (void)out_size; (void)ws_size;
  const float* x   = (const float*)d_in[0];
  const float* n1g = (const float*)d_in[1];
  const float* n1b = (const float*)d_in[2];
  const float* wq  = (const float*)d_in[3];
  const float* bq  = (const float*)d_in[4];
  const float* wk  = (const float*)d_in[5];
  const float* bk  = (const float*)d_in[6];
  const float* wv  = (const float*)d_in[7];
  const float* bv  = (const float*)d_in[8];
  const float* dqw = (const float*)d_in[9];
  const float* dqb = (const float*)d_in[10];
  const float* dkw = (const float*)d_in[11];
  const float* dkb = (const float*)d_in[12];
  const float* dvw = (const float*)d_in[13];
  const float* dvb = (const float*)d_in[14];
  const float* n2g = (const float*)d_in[15];
  const float* n2b = (const float*)d_in[16];
  const float* f1w = (const float*)d_in[17];
  const float* f1b = (const float*)d_in[18];
  const float* f2w = (const float*)d_in[19];
  const float* f2b = (const float*)d_in[20];
  float* y = (float*)d_out;

  // workspace carve-up (~276 MB)
  char* ws = (char*)d_ws;
  auto take = [&](size_t bytes) {
    char* p = ws; ws += (bytes + 255) & ~(size_t)255; return p;
  };
  unsigned short* xw   = (unsigned short*)take((size_t)NWIN * WROW * 2);
  unsigned short* qb   = (unsigned short*)take((size_t)NWIN * WROW * 2);
  unsigned short* kb   = (unsigned short*)take((size_t)NWIN * WROW * 2);
  unsigned short* vb   = (unsigned short*)take((size_t)NWIN * WROW * 2);
  float*          att  = (float*)take((size_t)NWIN * WROW * 4);
  float*          xres = (float*)take((size_t)VOL * C_ * 4);
  unsigned short* wqb  = (unsigned short*)take(C_ * C_ * 2);
  unsigned short* wkb  = (unsigned short*)take(C_ * C_ * 2);
  unsigned short* wvb  = (unsigned short*)take(C_ * C_ * 2);
  unsigned short* f1wb = (unsigned short*)take(HID * C_ * 2);
  unsigned short* f2wb = (unsigned short*)take(C_ * HID * 2);

  // weight conversions
  k_cvt_bf16<<<(C_ * C_ + 255) / 256, 256, 0, stream>>>(wq, wqb, C_ * C_);
  k_cvt_bf16<<<(C_ * C_ + 255) / 256, 256, 0, stream>>>(wk, wkb, C_ * C_);
  k_cvt_bf16<<<(C_ * C_ + 255) / 256, 256, 0, stream>>>(wv, wvb, C_ * C_);
  k_cvt_bf16<<<(HID * C_ + 255) / 256, 256, 0, stream>>>(f1w, f1wb, HID * C_);
  k_cvt_bf16<<<(C_ * HID + 255) / 256, 256, 0, stream>>>(f2w, f2wb, C_ * HID);

  k_ln1_window<<<NWIN * NTP / 8, 256, 0, stream>>>(x, n1g, n1b, xw);
  k_qkv<<<NWIN, 256, 0, stream>>>(xw, wqb, wkb, wvb, bq, bk, bv,
                                  dqw, dkw, dvw, dqb, dkb, dvb, qb, kb, vb);
  k_attn<<<NWIN * 22, 128, 0, stream>>>(qb, kb, vb, att);
  k_resid<<<(VOL * C_) / (256 * 4), 256, 0, stream>>>(x, att, xres);
  k_mlp<<<VOL / 64, 256, 0, stream>>>(xres, n2g, n2b, f1wb, f1b, f2wb, f2b, y);
}
